// Block_12841952215257
// MI455X (gfx1250) — matrix operand order repaired and checked by reading, not compiled, not run on hardware
//
#include <hip/hip_runtime.h>
#include <hip/hip_bf16.h>

// ---------------- types / helpers ----------------
typedef __attribute__((ext_vector_type(16))) __bf16 v16bf;
typedef __attribute__((ext_vector_type(8)))  float  v8f;
typedef __attribute__((ext_vector_type(4)))  unsigned int v4u;
typedef __attribute__((ext_vector_type(8)))  unsigned int v8u;

union FragAB { v16bf v; unsigned int u[8]; };
union FragC  { v8f  v; float f[8]; };

__device__ __forceinline__ unsigned short f2b(float f) {
  unsigned int x = __float_as_uint(f);
  x += 0x7FFFu + ((x >> 16) & 1u);          // round-to-nearest-even
  return (unsigned short)(x >> 16);
}
// A-matrix (16x32 bf16) per-VGPR K index, ISA 7.12.2:
//   K = (i>=4?16:0) + (i&3)*2 + 8*(lane>>4)
__device__ __forceinline__ int a_kidx(int i, int kh) { return ((i & 4) << 2) + ((i & 3) << 1) + (kh << 3); }
// B-matrix (32x16 bf16): lane col = lane&15, K = (i>=4?16:0) + (i&3)*2 + 8*(lane>>4)
__device__ __forceinline__ int b_kidx(int i, int kh) { return ((i & 4) << 2) + ((i & 3) << 1) + (kh << 3); }

// ---------------- TDM: 2-D tile load global -> LDS ----------------
// D# group0 (4 SGPRs) + group1 (8 SGPRs), VADDR2/3 = NULL (2-D tensor form).
// Fields per CDNA5 ISA. data_size = 2 bytes (bf16).
// pad_interval_code/pad_amount_code inject LDS row padding (bank-conflict avoidance).
__device__ __forceinline__ void tdm_load_2d(unsigned ldsOff, const unsigned short* gptr,
                                            unsigned tensor_d0, unsigned tensor_d1,
                                            unsigned tile_d0, unsigned tile_d1,
                                            unsigned stride_elems,
                                            unsigned pad_interval_code, unsigned pad_amount_code) {
  unsigned long long ga = (unsigned long long)(size_t)gptr;
  v4u g0;
  g0[0] = 1u;                                                   // count=1, user mode
  g0[1] = ldsOff;                                               // LDS byte address
  g0[2] = (unsigned)ga;                                         // global_addr[31:0]
  g0[3] = (unsigned)((ga >> 32) & 0x01FFFFFFu) | (2u << 30);    // addr[56:32] | type=2
  v8u g1;
  g1[0] = (1u << 16)                                            // data_size = 2B
        | (1u << 20)                                            // pad_enable
        | (pad_interval_code << 22) | (pad_amount_code << 25);
  g1[1] = (tensor_d0 & 0xFFFFu) << 16;                          // tensor_dim0[15:0]
  g1[2] = (tensor_d0 >> 16) | ((tensor_d1 & 0xFFFFu) << 16);    // dim0 hi | dim1 lo
  g1[3] = (tensor_d1 >> 16) | (tile_d0 << 16);                  // dim1 hi | tile_dim0
  g1[4] = tile_d1;                                              // tile_dim1 | tile_dim2=0
  g1[5] = stride_elems;                                         // tensor_dim0_stride[31:0]
  g1[6] = 0u;                                                   // stride hi | dim1_stride lo
  g1[7] = 0u;
  asm volatile("tensor_load_to_lds %0, %1" :: "s"(g0), "s"(g1) : "memory");
}

// ---------------- constants ----------------
static constexpr int C_EMBD = 1024;
static constexpr int N_HEAD = 16;
static constexpr int HEAD   = 64;
static constexpr int T_SEQ  = 2048;
static constexpr int BATCH  = 2;
static constexpr int ROWS   = BATCH * T_SEQ;  // 4096
static constexpr int QKV_N  = 3 * C_EMBD;     // 3072
static constexpr int FF_N   = 4 * C_EMBD;     // 4096

// ---------------- LayerNorm -> bf16 ----------------
__global__ __launch_bounds__(256) void ln_to_bf16(const float* __restrict__ x,
                                                  const float* __restrict__ g,
                                                  const float* __restrict__ b,
                                                  unsigned short* __restrict__ out) {
  const int row = blockIdx.x, tid = threadIdx.x;
  const float* xr = x + (size_t)row * C_EMBD;
  float v[4], s = 0.f, s2 = 0.f;
  #pragma unroll
  for (int i = 0; i < 4; ++i) { v[i] = xr[tid + 256 * i]; s += v[i]; s2 += v[i] * v[i]; }
  __shared__ float rs[256], rq[256];
  rs[tid] = s; rq[tid] = s2; __syncthreads();
  for (int st = 128; st > 0; st >>= 1) {
    if (tid < st) { rs[tid] += rs[tid + st]; rq[tid] += rq[tid + st]; }
    __syncthreads();
  }
  const float mu = rs[0] * (1.f / C_EMBD);
  const float var = rq[0] * (1.f / C_EMBD) - mu * mu;
  const float rstd = rsqrtf(var + 1e-5f);
  #pragma unroll
  for (int i = 0; i < 4; ++i) {
    int c = tid + 256 * i;
    out[(size_t)row * C_EMBD + c] = f2b((v[i] - mu) * rstd * g[c] + b[c]);
  }
}

// ---------------- weight conversion (fp32 -> bf16, TRANSPOSED to [N][K]) ----------------
// qkv: out[col][c], col in [0,3072) = {q|k|v} x head x d, c = embedding dim (K).
__global__ void pack_qkv_wT(const float* __restrict__ wq, const float* __restrict__ wk,
                            const float* __restrict__ wv, unsigned short* __restrict__ out) {
  size_t i = (size_t)blockIdx.x * 256 + threadIdx.x;     // i = col*1024 + c
  if (i >= (size_t)C_EMBD * QKV_N) return;
  int col = (int)(i / C_EMBD), c = (int)(i % C_EMBD);
  int sec = col >> 10, rest = col & 1023, h = rest >> 6, d = rest & 63;
  const float* w = (sec == 0) ? wq : ((sec == 1) ? wk : wv);
  out[i] = f2b(w[((size_t)h * C_EMBD + c) * HEAD + d]);
}

// generic transpose-convert: in [K][N] fp32 -> out [N][K] bf16
__global__ void conv_bf16_T(const float* __restrict__ in, unsigned short* __restrict__ out,
                            int K, int N) {
  size_t total = (size_t)K * N;
  for (size_t i = (size_t)blockIdx.x * 256 + threadIdx.x; i < total; i += (size_t)gridDim.x * 256) {
    int n = (int)(i / K), k = (int)(i % K);
    out[i] = f2b(in[(size_t)k * N + n]);
  }
}

// ---------------- TDM double-buffered WMMA GEMM ----------------
// C[M,N] = A[M,K] * Bt[N,K]^T.  Block tile 128x128, K-step 64, 8 waves,
// wave tile 64x32 -> 16 WMMAs per K-step. Tiles staged by tensor_load_to_lds
// (wave 0 issues; TENSORcnt + one barrier per step), double buffered.
template<bool OUTF, bool OUTB, bool BIAS, bool RELU, bool RESID>
__global__ __launch_bounds__(256) void gemm_bf16_wmma(
    const unsigned short* __restrict__ A, const unsigned short* __restrict__ Bt,
    const float* __restrict__ bias, const float* __restrict__ resid,
    float* __restrict__ outF, unsigned short* __restrict__ outB,
    int M, int N, int K) {
  // 64 data elems + 8 pad elems per row (128B data + 16B pad = TDM pad 4 dwords / 32 dwords)
  __shared__ unsigned short As[2][128][72];
  __shared__ unsigned short Bs[2][128][72];
  const int tid = threadIdx.x;
  const int lane = tid & 31, w = tid >> 5;
  const int m16 = lane & 15, kh = lane >> 4;
  const int wm = w >> 2, wn = w & 3;
  const int rowBase = blockIdx.y * 128;
  const int colBase = blockIdx.x * 128;
  const int nIter = K >> 6;

  const unsigned short* Ab = A  + (size_t)rowBase * K;
  const unsigned short* Bb = Bt + (size_t)colBase * K;

  if (w == 0) {   // prologue: fill buffer 0
    tdm_load_2d((unsigned)(size_t)&As[0][0][0], Ab, (unsigned)K, (unsigned)M, 64u, 128u, (unsigned)K, 4u, 3u);
    tdm_load_2d((unsigned)(size_t)&Bs[0][0][0], Bb, (unsigned)K, (unsigned)N, 64u, 128u, (unsigned)K, 4u, 3u);
  }

  FragC acc[4][2];
  #pragma unroll
  for (int mt = 0; mt < 4; ++mt)
    #pragma unroll
    for (int nt = 0; nt < 2; ++nt)
      #pragma unroll
      for (int j = 0; j < 8; ++j) acc[mt][nt].f[j] = 0.f;

  for (int it = 0; it < nIter; ++it) {
    if (w == 0) __builtin_amdgcn_s_wait_tensorcnt(0);   // buf[it&1] resident in LDS
    __syncthreads();                                    // publish; prior compute done
    if (w == 0 && it + 1 < nIter) {                     // prefetch next K-slab
      const int nb = (it + 1) & 1, k0n = (it + 1) << 6;
      tdm_load_2d((unsigned)(size_t)&As[nb][0][0], Ab + k0n, (unsigned)K, (unsigned)M, 64u, 128u, (unsigned)K, 4u, 3u);
      tdm_load_2d((unsigned)(size_t)&Bs[nb][0][0], Bb + k0n, (unsigned)K, (unsigned)N, 64u, 128u, (unsigned)K, 4u, 3u);
    }
    const int cur = it & 1;
    #pragma unroll
    for (int kc = 0; kc < 2; ++kc) {
      FragAB af[4], bfm[2];
      #pragma unroll
      for (int mt = 0; mt < 4; ++mt)
        #pragma unroll
        for (int i = 0; i < 8; ++i)
          af[mt].u[i] = *(const unsigned int*)&As[cur][wm * 64 + mt * 16 + m16][kc * 32 + a_kidx(i, kh)];
      #pragma unroll
      for (int nt = 0; nt < 2; ++nt)
        #pragma unroll
        for (int i = 0; i < 8; ++i)
          bfm[nt].u[i] = *(const unsigned int*)&Bs[cur][wn * 32 + nt * 16 + m16][kc * 32 + b_kidx(i, kh)];
      #pragma unroll
      for (int mt = 0; mt < 4; ++mt)
        #pragma unroll
        for (int nt = 0; nt < 2; ++nt)
          acc[mt][nt].v = __builtin_amdgcn_wmma_f32_16x16x32_bf16(
              false, af[mt].v, false, bfm[nt].v, (short)0, acc[mt][nt].v, false, false);
    }
    // no trailing barrier needed: buffer `cur` is only overwritten by the DMA
    // issued after barrier(it+1), which every wave reaches only after this compute.
  }

  #pragma unroll
  for (int mt = 0; mt < 4; ++mt)
    #pragma unroll
    for (int nt = 0; nt < 2; ++nt) {
      const int gc = colBase + wn * 32 + nt * 16 + m16;
      const float bb = BIAS ? bias[gc] : 0.f;
      #pragma unroll
      for (int j = 0; j < 8; ++j) {
        const int gr = rowBase + wm * 64 + mt * 16 + j + 8 * kh;
        float vl = acc[mt][nt].f[j];
        if (BIAS)  vl += bb;
        if (RELU)  vl = fmaxf(vl, 0.f);
        if (RESID) vl += resid[(size_t)gr * N + gc];
        if (OUTF)  outF[(size_t)gr * N + gc] = vl;
        if (OUTB)  outB[(size_t)gr * N + gc] = f2b(vl);
      }
    }
}

// ---------------- fused causal flash attention ----------------
// grid (T/64, H, B), 128 threads = 4 waves; wave handles 16 query rows, D=64.
__global__ __launch_bounds__(128) void attn_fused(const unsigned short* __restrict__ qkv,
                                                  unsigned short* __restrict__ out) {
  __shared__ unsigned short Ks[32][72];      // [key][d]
  __shared__ unsigned short Vt[64][40];      // [d][key] (transposed)
  __shared__ unsigned short Ps[4][16][40];   // per-wave P tile, [qrow][key]
  const int tid = threadIdx.x;
  const int lane = tid & 31, w = tid >> 5;
  const int m16 = lane & 15, kh = lane >> 4;
  const int qbase = blockIdx.x * 64;
  const int h = blockIdx.y, bb = blockIdx.z;
  const int rowOff = bb * T_SEQ;
  const int qrow0 = qbase + w * 16;
  const int qmaxw = qrow0 + 15;
  const int hq = h * HEAD, hk = C_EMBD + h * HEAD, hv = 2 * C_EMBD + h * HEAD;

  FragAB aq[2];                               // Q (16x64) as two K=32 A-fragments
  {
    const unsigned short* qp = qkv + (size_t)(rowOff + qrow0 + m16) * QKV_N + hq;
    #pragma unroll
    for (int kc = 0; kc < 2; ++kc)
      #pragma unroll
      for (int i = 0; i < 8; ++i)
        aq[kc].u[i] = *(const unsigned int*)(qp + kc * 32 + a_kidx(i, kh));
  }
  float mrow[8], lrow[8];
  FragC o[4];
  #pragma unroll
  for (int j = 0; j < 8; ++j) { mrow[j] = -1e30f; lrow[j] = 0.f; }
  #pragma unroll
  for (int nt = 0; nt < 4; ++nt)
    #pragma unroll
    for (int j = 0; j < 8; ++j) o[nt].f[j] = 0.f;

  const int smax = qbase + 63;
  for (int s0 = 0; s0 <= smax; s0 += 32) {
    #pragma unroll
    for (int half = 0; half < 2; ++half) {   // load K chunk + V chunk (32x64 each)
      int idx = tid * 8 + half * 1024;
      int r = idx >> 6, c = idx & 63;
      size_t rowp = (size_t)(rowOff + s0 + r) * QKV_N;
      uint4 kv4 = *(const uint4*)(qkv + rowp + hk + c);
      *(uint4*)&Ks[r][c] = kv4;
      uint4 vv4 = *(const uint4*)(qkv + rowp + hv + c);
      const unsigned short* vp = (const unsigned short*)&vv4;
      #pragma unroll
      for (int j = 0; j < 8; ++j) Vt[c + j][r] = vp[j];
    }
    __syncthreads();
    const bool active = (s0 <= qmaxw);       // wave-uniform: EXEC all-ones at WMMA
    if (active) {
      FragC S[2];
      #pragma unroll
      for (int sc = 0; sc < 2; ++sc) {
        #pragma unroll
        for (int j = 0; j < 8; ++j) S[sc].f[j] = 0.f;
        #pragma unroll
        for (int kc = 0; kc < 2; ++kc) {
          FragAB bk;                           // B = K^T: B[d][key] = Ks[key][d]
          #pragma unroll
          for (int i = 0; i < 8; ++i)
            bk.u[i] = *(const unsigned int*)&Ks[sc * 16 + m16][kc * 32 + b_kidx(i, kh)];
          S[sc].v = __builtin_amdgcn_wmma_f32_16x16x32_bf16(
              false, aq[kc].v, false, bk.v, (short)0, S[sc].v, false, false);
        }
      }
      #pragma unroll
      for (int sc = 0; sc < 2; ++sc) {        // scale + causal mask
        const int key = s0 + sc * 16 + m16;
        #pragma unroll
        for (int j = 0; j < 8; ++j) {
          const int qg = qrow0 + j + 8 * kh;
          float sv = S[sc].f[j] * 0.125f;     // 1/sqrt(64)
          if (key > qg) sv = -1e30f;
          S[sc].f[j] = sv;
        }
      }
      #pragma unroll
      for (int j = 0; j < 8; ++j) {           // online softmax, half-wave reductions
        float cm = fmaxf(S[0].f[j], S[1].f[j]);
        #pragma unroll
        for (int d = 1; d < 16; d <<= 1) cm = fmaxf(cm, __shfl_xor(cm, d, 32));
        const float mnew = fmaxf(mrow[j], cm);
        const float alpha = __expf(mrow[j] - mnew);
        const float p0 = __expf(S[0].f[j] - mnew);
        const float p1 = __expf(S[1].f[j] - mnew);
        float ps = p0 + p1;
        #pragma unroll
        for (int d = 1; d < 16; d <<= 1) ps += __shfl_xor(ps, d, 32);
        mrow[j] = mnew;
        lrow[j] = lrow[j] * alpha + ps;
        #pragma unroll
        for (int nt = 0; nt < 4; ++nt) o[nt].f[j] *= alpha;
        const int prow = j + 8 * kh;          // C-layout row -> LDS for relayout
        Ps[w][prow][m16] = f2b(p0);
        Ps[w][prow][16 + m16] = f2b(p1);
      }
    }
    __syncthreads();
    if (active) {                              // O += P (16x32) @ V (32x64)
      FragAB pf;
      #pragma unroll
      for (int i = 0; i < 8; ++i)
        pf.u[i] = *(const unsigned int*)&Ps[w][m16][a_kidx(i, kh)];
      #pragma unroll
      for (int nt = 0; nt < 4; ++nt) {
        FragAB vf;
        #pragma unroll
        for (int i = 0; i < 8; ++i)
          vf.u[i] = *(const unsigned int*)&Vt[nt * 16 + m16][b_kidx(i, kh)];
        o[nt].v = __builtin_amdgcn_wmma_f32_16x16x32_bf16(
            false, pf.v, false, vf.v, (short)0, o[nt].v, false, false);
      }
    }
    __syncthreads();
  }
  #pragma unroll
  for (int nt = 0; nt < 4; ++nt)
    #pragma unroll
    for (int j = 0; j < 8; ++j) {
      const int qg = qrow0 + j + 8 * kh;
      out[(size_t)(rowOff + qg) * C_EMBD + h * HEAD + nt * 16 + m16] = f2b(o[nt].f[j] / lrow[j]);
    }
}

// ---------------- launch ----------------
extern "C" void kernel_launch(void* const* d_in, const int* in_sizes, int n_in,
                              void* d_out, int out_size, void* d_ws, size_t ws_size,
                              hipStream_t stream) {
  (void)in_sizes; (void)n_in; (void)out_size; (void)ws_size;
  const float* x      = (const float*)d_in[0];
  const float* wq     = (const float*)d_in[1];
  const float* wk     = (const float*)d_in[2];
  const float* wv     = (const float*)d_in[3];
  const float* w_proj = (const float*)d_in[4];
  const float* b_proj = (const float*)d_in[5];
  const float* w1     = (const float*)d_in[6];
  const float* b1     = (const float*)d_in[7];
  const float* w2     = (const float*)d_in[8];
  const float* b2     = (const float*)d_in[9];
  const float* g1     = (const float*)d_in[10];
  const float* be1    = (const float*)d_in[11];
  const float* g2     = (const float*)d_in[12];
  const float* be2    = (const float*)d_in[13];

  char* ws = (char*)d_ws;
  size_t off = 0;
  auto alloc = [&](size_t bytes) { void* p = ws + off; off += (bytes + 255) & ~(size_t)255; return p; };
  unsigned short* h1    = (unsigned short*)alloc((size_t)ROWS * C_EMBD * 2);   // LN1(x) bf16
  unsigned short* wqkvT = (unsigned short*)alloc((size_t)C_EMBD * QKV_N * 2);  // [3072][1024] bf16
  unsigned short* qkvB  = (unsigned short*)alloc((size_t)ROWS * QKV_N * 2);    // q|k|v bf16
  unsigned short* attnB = (unsigned short*)alloc((size_t)ROWS * C_EMBD * 2);   // attn concat bf16
  unsigned short* wprjT = (unsigned short*)alloc((size_t)C_EMBD * C_EMBD * 2); // [1024][1024]^T
  float*          x1    = (float*)alloc((size_t)ROWS * C_EMBD * 4);            // x + sa (f32)
  unsigned short* h2    = (unsigned short*)alloc((size_t)ROWS * C_EMBD * 2);   // LN2(x1) bf16
  unsigned short* w1T   = (unsigned short*)alloc((size_t)C_EMBD * FF_N * 2);   // [4096][1024]
  unsigned short* ff1   = (unsigned short*)alloc((size_t)ROWS * FF_N * 2);     // relu(.) bf16
  unsigned short* w2T   = (unsigned short*)alloc((size_t)FF_N * C_EMBD * 2);   // [1024][4096]

  // weight conversions (transposed to [N][K] for TDM tile loads)
  pack_qkv_wT<<<((size_t)C_EMBD * QKV_N + 255) / 256, 256, 0, stream>>>(wq, wk, wv, wqkvT);
  conv_bf16_T<<<1024, 256, 0, stream>>>(w_proj, wprjT, C_EMBD, C_EMBD);
  conv_bf16_T<<<2048, 256, 0, stream>>>(w1, w1T, C_EMBD, FF_N);
  conv_bf16_T<<<2048, 256, 0, stream>>>(w2, w2T, FF_N, C_EMBD);

  // attention path
  ln_to_bf16<<<ROWS, 256, 0, stream>>>(x, g1, be1, h1);
  gemm_bf16_wmma<false, true, false, false, false>
      <<<dim3(QKV_N / 128, ROWS / 128), 256, 0, stream>>>(
          h1, wqkvT, nullptr, nullptr, nullptr, qkvB, ROWS, QKV_N, C_EMBD);
  attn_fused<<<dim3(T_SEQ / 64, N_HEAD, BATCH), 128, 0, stream>>>(qkvB, attnB);
  gemm_bf16_wmma<true, false, true, false, true>
      <<<dim3(C_EMBD / 128, ROWS / 128), 256, 0, stream>>>(
          attnB, wprjT, b_proj, x, x1, nullptr, ROWS, C_EMBD, C_EMBD);

  // FFN path
  ln_to_bf16<<<ROWS, 256, 0, stream>>>(x1, g2, be2, h2);
  gemm_bf16_wmma<false, true, true, true, false>
      <<<dim3(FF_N / 128, ROWS / 128), 256, 0, stream>>>(
          h2, w1T, b1, nullptr, nullptr, ff1, ROWS, FF_N, C_EMBD);
  gemm_bf16_wmma<true, false, true, false, true>
      <<<dim3(C_EMBD / 128, ROWS / 128), 256, 0, stream>>>(
          ff1, w2T, b2, x1, (float*)d_out, nullptr, ROWS, C_EMBD, FF_N);
}
